// SpeechAwareTransformer_15384572854630
// MI455X (gfx1250) — matrix operand order repaired and checked by reading, not compiled, not run on hardware
//
#include <hip/hip_runtime.h>
#include <hip/hip_bf16.h>
#include <math.h>

// Problem constants
#define S_     1024
#define B_     8
#define D_     512
#define H_     8
#define HD_    64
#define NHID_  2048
#define L_     4
#define NTOK_  10000
#define NTOKP_ 10240
#define M_     8192   // S_*B_

typedef _Float16 v16h __attribute__((ext_vector_type(16)));
typedef _Float16 v8h  __attribute__((ext_vector_type(8)));
typedef float    v8f  __attribute__((ext_vector_type(8)));

__device__ __forceinline__ float sigm_(float x) { return 1.f / (1.f + expf(-x)); }

// Load a 16x32 f16 A-fragment (row-major source, ld = lda halves).
// Lane L holds M=L%16; halves j=0..15 -> K = k0 + (L/16)*16 + j (matches B). One contiguous 32B load.
__device__ __forceinline__ v16h ldA_(const _Float16* __restrict__ row, int k0, int lh) {
  return *(const v16h*)(row + k0 + lh * 16);
}

// -------------------------------------------------------------------------
// Generic WMMA GEMM: C[M, ldc] = A[M,K](f16) * W[N,K](f16)^T + bias
// Grid: (M/128, Npad/256), 256 threads (8 waves), wave tile 64x64 (4x4 frags)
// -------------------------------------------------------------------------
__global__ __launch_bounds__(256) void k_gemm(
    const _Float16* __restrict__ A, const _Float16* __restrict__ W,
    const float* __restrict__ bias, float* __restrict__ C,
    int K, int ldc, int Nvalid)
{
  const int lane = threadIdx.x & 31;
  const int wv   = threadIdx.x >> 5;
  const int wm   = wv >> 2, wn = wv & 3;
  const int m0 = blockIdx.x * 128 + wm * 64;
  const int n0 = blockIdx.y * 256 + wn * 64;
  const int lh = lane >> 4;
  const int ln = lane & 15;

  v8f acc[4][4];
#pragma unroll
  for (int i = 0; i < 4; ++i)
#pragma unroll
    for (int j = 0; j < 4; ++j) acc[i][j] = {};

  const _Float16* arow[4];
  const _Float16* brow[4];
#pragma unroll
  for (int i = 0; i < 4; ++i) arow[i] = A + (size_t)(m0 + 16 * i + ln) * K;
#pragma unroll
  for (int j = 0; j < 4; ++j) brow[j] = W + (size_t)(n0 + 16 * j + ln) * K + lh * 16;

  for (int kt = 0; kt < K; kt += 32) {
    v16h a[4], b[4];
#pragma unroll
    for (int i = 0; i < 4; ++i) a[i] = ldA_(arow[i], kt, lh);
#pragma unroll
    for (int j = 0; j < 4; ++j) b[j] = *(const v16h*)(brow[j] + kt);
#pragma unroll
    for (int i = 0; i < 4; ++i)
#pragma unroll
      for (int j = 0; j < 4; ++j)
        acc[i][j] = __builtin_amdgcn_wmma_f32_16x16x32_f16(
            false, a[i], false, b[j], (short)0, acc[i][j], false, false);
  }

#pragma unroll
  for (int j = 0; j < 4; ++j) {
    const int col = n0 + 16 * j + ln;
    if (col < Nvalid) {
      const float bb = bias[col];
#pragma unroll
      for (int i = 0; i < 4; ++i) {
        const int rbase = m0 + 16 * i + 8 * lh;
#pragma unroll
        for (int r = 0; r < 8; ++r)
          C[(size_t)(rbase + r) * ldc + col] = acc[i][j][r] + bb;
      }
    }
  }
}

// -------------------------------------------------------------------------
// Persistent LSTM scan: one WG, 1024 threads (32 waves).
// zin[t*B+b, 4D] = x_t @ W_ih^T + b_ih + b_hh (precomputed by k_gemm).
// Each step: zbuf = h @ W_hh^T via WMMA (h in LDS as 16x512 f16, rows 8..15 = 0),
// then gate update, h/c in LDS, h also written to global (f32 + f16).
// -------------------------------------------------------------------------
__global__ __launch_bounds__(1024) void k_lstm(
    const float* __restrict__ zin, const _Float16* __restrict__ whh,
    float* __restrict__ xout, _Float16* __restrict__ xout16)
{
  extern __shared__ char smem[];
  _Float16* hbuf = (_Float16*)smem;                              // 16*512 f16 = 16KB
  float*    cbuf = (float*)(smem + 16 * 512 * 2);                // 8*512  f32 = 16KB
  float*    zbuf = (float*)(smem + 16 * 512 * 2 + 8 * 512 * 4);  // 8*2048 f32 = 64KB

  const int tid  = threadIdx.x;
  const int lane = tid & 31;
  const int wv   = tid >> 5;
  const int lh   = lane >> 4;
  const int ln   = lane & 15;

  for (int i = tid; i < 16 * 512; i += 1024) hbuf[i] = (_Float16)0.f;
  for (int i = tid; i < 8 * 512;  i += 1024) cbuf[i] = 0.f;
  __syncthreads();

  for (int t = 0; t < S_; ++t) {
#pragma unroll
    for (int j = 0; j < 4; ++j) {
      const int n0 = (wv * 4 + j) * 16;
      v8f acc = {};
      const _Float16* hrow = hbuf + (size_t)ln * 512;
      const _Float16* brow = whh + (size_t)(n0 + ln) * 512 + lh * 16;
      for (int kt = 0; kt < 512; kt += 32) {
        v16h a = ldA_(hrow, kt, lh);
        v16h b = *(const v16h*)(brow + kt);
        acc = __builtin_amdgcn_wmma_f32_16x16x32_f16(
            false, a, false, b, (short)0, acc, false, false);
      }
      if (lh == 0) {  // lanes 0..15 hold M=0..7 (valid batch rows)
#pragma unroll
        for (int r = 0; r < 8; ++r) zbuf[r * 2048 + n0 + ln] = acc[r];
      }
    }
    __syncthreads();

    const float* zr = zin + (size_t)t * 8 * 2048;
#pragma unroll
    for (int it = 0; it < 4; ++it) {
      const int idx = tid + it * 1024;           // 0..4095
      const int bb = idx >> 9, n = idx & 511;
      const float zi = zbuf[bb * 2048 + n]          + zr[bb * 2048 + n];
      const float zf = zbuf[bb * 2048 + 512 + n]    + zr[bb * 2048 + 512 + n];
      const float zg = zbuf[bb * 2048 + 1024 + n]   + zr[bb * 2048 + 1024 + n];
      const float zo = zbuf[bb * 2048 + 1536 + n]   + zr[bb * 2048 + 1536 + n];
      const float c  = sigm_(zf) * cbuf[bb * 512 + n] + sigm_(zi) * tanhf(zg);
      const float h  = sigm_(zo) * tanhf(c);
      cbuf[bb * 512 + n] = c;
      hbuf[bb * 512 + n] = (_Float16)h;
      xout  [(size_t)(t * 8 + bb) * 512 + n] = h;
      xout16[(size_t)(t * 8 + bb) * 512 + n] = (_Float16)h;
    }
    __syncthreads();
  }
}

// -------------------------------------------------------------------------
// Banded attention (|i-j| <= 32), one wave per (b,h,s). Online softmax.
// q/k/v rows: [(s*B+b)*D + h*64 + d], fp32. Output f16 (feeds O-proj GEMM).
// -------------------------------------------------------------------------
__global__ __launch_bounds__(256) void k_attn(
    const float* __restrict__ q, const float* __restrict__ k,
    const float* __restrict__ v, _Float16* __restrict__ out)
{
  const int gw   = (blockIdx.x * 256 + threadIdx.x) >> 5;  // 0..65535
  const int lane = threadIdx.x & 31;
  const int s  = gw & (S_ - 1);
  const int bh = gw >> 10;
  const int b  = bh >> 3, h = bh & 7;
  const size_t hoff = (size_t)h * 64 + lane * 2;

  const float* qp = q + (size_t)(s * B_ + b) * D_ + hoff;
  const float q0 = qp[0], q1 = qp[1];

  float m = -INFINITY, den = 0.f, a0 = 0.f, a1 = 0.f;
  const int j0 = (s - 32 < 0) ? 0 : s - 32;
  const int j1 = (s + 32 > S_ - 1) ? S_ - 1 : s + 32;
  for (int j = j0; j <= j1; ++j) {
    const size_t rb = (size_t)(j * B_ + b) * D_ + hoff;
    const float* kp = k + rb;
    float p = q0 * kp[0] + q1 * kp[1];
#pragma unroll
    for (int off = 16; off; off >>= 1) p += __shfl_xor(p, off, 32);
    p *= 0.125f;  // 1/sqrt(64)
    const float nm   = fmaxf(m, p);
    const float corr = expf(m - nm);
    const float e    = expf(p - nm);
    const float* vp  = v + rb;
    a0 = a0 * corr + e * vp[0];
    a1 = a1 * corr + e * vp[1];
    den = den * corr + e;
    m = nm;
  }
  const float inv = 1.f / den;
  _Float16* op = out + (size_t)(s * B_ + b) * D_ + hoff;
  op[0] = (_Float16)(a0 * inv);
  op[1] = (_Float16)(a1 * inv);
}

// -------------------------------------------------------------------------
// y = LayerNorm(x + r) * g + be ; writes f32 (residual) + f16 (GEMM input).
// One 256-thread block per row of D=512 (2 elems/thread). In-place safe.
// -------------------------------------------------------------------------
__global__ __launch_bounds__(256) void k_add_ln(
    const float* __restrict__ x, const float* __restrict__ r,
    const float* __restrict__ g, const float* __restrict__ be,
    float* __restrict__ yo, _Float16* __restrict__ yo16)
{
  __shared__ float red[16];
  const size_t base = (size_t)blockIdx.x * D_;
  const int tid = threadIdx.x;
  const float v0 = x[base + tid]       + r[base + tid];
  const float v1 = x[base + 256 + tid] + r[base + 256 + tid];
  float s  = v0 + v1;
  float s2 = v0 * v0 + v1 * v1;
#pragma unroll
  for (int off = 16; off; off >>= 1) { s += __shfl_xor(s, off, 32); s2 += __shfl_xor(s2, off, 32); }
  if ((tid & 31) == 0) { red[tid >> 5] = s; red[8 + (tid >> 5)] = s2; }
  __syncthreads();
  float ts = 0.f, ts2 = 0.f;
#pragma unroll
  for (int i = 0; i < 8; ++i) { ts += red[i]; ts2 += red[8 + i]; }
  const float mean = ts * (1.f / 512.f);
  const float var  = ts2 * (1.f / 512.f) - mean * mean;
  const float rstd = rsqrtf(var + 1e-5f);
  const float y0 = (v0 - mean) * rstd * g[tid] + be[tid];
  const float y1 = (v1 - mean) * rstd * g[256 + tid] + be[256 + tid];
  yo[base + tid] = y0;           yo[base + 256 + tid] = y1;
  yo16[base + tid] = (_Float16)y0; yo16[base + 256 + tid] = (_Float16)y1;
}

// Exact (erf) GELU, f32 -> f16
__global__ void k_gelu(const float* __restrict__ in, _Float16* __restrict__ out, int n)
{
  for (int i = blockIdx.x * blockDim.x + threadIdx.x; i < n; i += gridDim.x * blockDim.x) {
    const float x = in[i];
    out[i] = (_Float16)(0.5f * x * (1.f + erff(x * 0.70710678118654752f)));
  }
}

__global__ void k_cvt(const float* __restrict__ in, _Float16* __restrict__ out, int n)
{
  for (int i = blockIdx.x * blockDim.x + threadIdx.x; i < n; i += gridDim.x * blockDim.x)
    out[i] = (_Float16)in[i];
}

__global__ void k_fill16(_Float16* __restrict__ p, int n)
{
  for (int i = blockIdx.x * blockDim.x + threadIdx.x; i < n; i += gridDim.x * blockDim.x)
    p[i] = (_Float16)0.f;
}

__global__ void k_padbias(const float* __restrict__ b, float* __restrict__ bp)
{
  const int i = blockIdx.x * blockDim.x + threadIdx.x;
  if (i < NTOKP_) bp[i] = (i < NTOK_) ? b[i] : 0.f;
}

__global__ void k_bsum(const float* __restrict__ a, const float* __restrict__ b,
                       float* __restrict__ o)
{
  const int i = blockIdx.x * blockDim.x + threadIdx.x;
  if (i < 4 * D_) o[i] = a[i] + b[i];
}

// x0 = emb[src]*sqrt(D) + pos_emb[s]  ->  f16 (feeds LSTM input-proj GEMM)
__global__ void k_embed(const int* __restrict__ src, const float* __restrict__ emb,
                        const float* __restrict__ pos, _Float16* __restrict__ x16, int n)
{
  for (int i = blockIdx.x * blockDim.x + threadIdx.x; i < n; i += gridDim.x * blockDim.x) {
    const int d = i & 511;
    const int sb = i >> 9;       // s*B+b
    const int s = sb >> 3;
    const int tok = src[sb];
    const float val = emb[(size_t)tok * D_ + d] * 22.627416997969522f + pos[(size_t)s * D_ + d];
    x16[i] = (_Float16)val;
  }
}

// In-place row-wise log_softmax over NTOK_ columns (block per row)
__global__ __launch_bounds__(256) void k_logsoftmax(float* __restrict__ logits)
{
  __shared__ float red[8];
  float* p = logits + (size_t)blockIdx.x * NTOK_;
  const int tid = threadIdx.x;

  float mx = -INFINITY;
  for (int i = tid; i < NTOK_; i += 256) mx = fmaxf(mx, p[i]);
#pragma unroll
  for (int off = 16; off; off >>= 1) mx = fmaxf(mx, __shfl_xor(mx, off, 32));
  if ((tid & 31) == 0) red[tid >> 5] = mx;
  __syncthreads();
  float bm = -INFINITY;
#pragma unroll
  for (int i = 0; i < 8; ++i) bm = fmaxf(bm, red[i]);
  __syncthreads();

  float sum = 0.f;
  for (int i = tid; i < NTOK_; i += 256) sum += expf(p[i] - bm);
#pragma unroll
  for (int off = 16; off; off >>= 1) sum += __shfl_xor(sum, off, 32);
  if ((tid & 31) == 0) red[tid >> 5] = sum;
  __syncthreads();
  float bs = 0.f;
#pragma unroll
  for (int i = 0; i < 8; ++i) bs += red[i];
  const float lg = bm + logf(bs);
  for (int i = tid; i < NTOK_; i += 256) p[i] = p[i] - lg;
}

// =========================================================================
extern "C" void kernel_launch(void* const* d_in, const int* in_sizes, int n_in,
                              void* d_out, int out_size, void* d_ws, size_t ws_size,
                              hipStream_t stream)
{
  (void)in_sizes; (void)n_in; (void)out_size; (void)ws_size;

  const int*   src  = (const int*)  d_in[0];
  const float* emb  = (const float*)d_in[1];
  const float* pos  = (const float*)d_in[2];
  const float* wih  = (const float*)d_in[3];
  const float* whh  = (const float*)d_in[4];
  const float* bih  = (const float*)d_in[5];
  const float* bhh  = (const float*)d_in[6];
  const float* qw   = (const float*)d_in[7];
  const float* qb   = (const float*)d_in[8];
  const float* kw   = (const float*)d_in[9];
  const float* kb   = (const float*)d_in[10];
  const float* vw   = (const float*)d_in[11];
  const float* vb   = (const float*)d_in[12];
  const float* ow   = (const float*)d_in[13];
  const float* ob   = (const float*)d_in[14];
  const float* w1   = (const float*)d_in[15];
  const float* b1   = (const float*)d_in[16];
  const float* w2   = (const float*)d_in[17];
  const float* b2   = (const float*)d_in[18];
  const float* g1   = (const float*)d_in[19];
  const float* be1  = (const float*)d_in[20];
  const float* g2   = (const float*)d_in[21];
  const float* be2  = (const float*)d_in[22];
  const float* decw = (const float*)d_in[23];
  const float* decb = (const float*)d_in[24];
  float* out = (float*)d_out;

  // ---- workspace carve-out ----
  char* ws = (char*)d_ws;
  size_t off = 0;
  auto alloc = [&](size_t bytes) -> char* {
    char* p = ws + off;
    off += (bytes + 255) & ~(size_t)255;
    return p;
  };
  _Float16* wih16  = (_Float16*)alloc((size_t)2048 * 512 * 2);
  _Float16* whh16  = (_Float16*)alloc((size_t)2048 * 512 * 2);
  _Float16* qw16   = (_Float16*)alloc((size_t)L_ * 512 * 512 * 2);
  _Float16* kw16   = (_Float16*)alloc((size_t)L_ * 512 * 512 * 2);
  _Float16* vw16   = (_Float16*)alloc((size_t)L_ * 512 * 512 * 2);
  _Float16* ow16   = (_Float16*)alloc((size_t)L_ * 512 * 512 * 2);
  _Float16* w116   = (_Float16*)alloc((size_t)L_ * 2048 * 512 * 2);
  _Float16* w216   = (_Float16*)alloc((size_t)L_ * 512 * 2048 * 2);
  _Float16* dec16  = (_Float16*)alloc((size_t)NTOKP_ * 512 * 2);
  float*    decbp  = (float*)   alloc((size_t)NTOKP_ * 4);
  float*    bsum   = (float*)   alloc((size_t)2048 * 4);
  _Float16* x0f16  = (_Float16*)alloc((size_t)M_ * 512 * 2);
  float*    zin    = (float*)   alloc((size_t)M_ * 2048 * 4);   // also FFN mid f32
  float*    xf32   = (float*)   alloc((size_t)M_ * 512 * 4);
  _Float16* xf16   = (_Float16*)alloc((size_t)M_ * 512 * 2);
  float*    qf     = (float*)   alloc((size_t)M_ * 512 * 4);
  float*    kf     = (float*)   alloc((size_t)M_ * 512 * 4);
  float*    vf     = (float*)   alloc((size_t)M_ * 512 * 4);
  _Float16* attn16 = (_Float16*)alloc((size_t)M_ * 512 * 2);
  float*    t0     = (float*)   alloc((size_t)M_ * 512 * 4);
  _Float16* hm16   = (_Float16*)alloc((size_t)M_ * 2048 * 2);

  auto cvt = [&](const float* s_, _Float16* d_, int n) {
    k_cvt<<<dim3((n + 255) / 256), dim3(256), 0, stream>>>(s_, d_, n);
  };
  auto gemm = [&](const _Float16* A, const _Float16* W, const float* bias, float* C,
                  int Npad, int Nvalid, int K, int ldc) {
    k_gemm<<<dim3(M_ / 128, Npad / 256), dim3(256), 0, stream>>>(A, W, bias, C, K, ldc, Nvalid);
  };

  // ---- weight conversion to f16 (weights stay hot in 192MB L2) ----
  cvt(wih, wih16, 2048 * 512);
  cvt(whh, whh16, 2048 * 512);
  cvt(qw, qw16, L_ * 512 * 512);
  cvt(kw, kw16, L_ * 512 * 512);
  cvt(vw, vw16, L_ * 512 * 512);
  cvt(ow, ow16, L_ * 512 * 512);
  cvt(w1, w116, L_ * 2048 * 512);
  cvt(w2, w216, L_ * 512 * 2048);
  cvt(decw, dec16, NTOK_ * 512);
  k_fill16<<<dim3((240 * 512 + 255) / 256), dim3(256), 0, stream>>>(
      dec16 + (size_t)NTOK_ * 512, 240 * 512);
  k_padbias<<<dim3(NTOKP_ / 256), dim3(256), 0, stream>>>(decb, decbp);
  k_bsum<<<dim3(8), dim3(256), 0, stream>>>(bih, bhh, bsum);

  // ---- embedding -> f16 ----
  k_embed<<<dim3(16384), dim3(256), 0, stream>>>(src, emb, pos, x0f16, M_ * 512);

  // ---- LSTM: input projection (big WMMA GEMM), then persistent scan ----
  gemm(x0f16, wih16, bsum, zin, 2048, 2048, 512, 2048);
  k_lstm<<<dim3(1), dim3(1024), 98304, stream>>>(zin, whh16, xf32, xf16);

  // ---- transformer layers ----
  for (int l = 0; l < L_; ++l) {
    gemm(xf16, qw16 + (size_t)l * 512 * 512, qb + l * 512, qf, 512, 512, 512, 512);
    gemm(xf16, kw16 + (size_t)l * 512 * 512, kb + l * 512, kf, 512, 512, 512, 512);
    gemm(xf16, vw16 + (size_t)l * 512 * 512, vb + l * 512, vf, 512, 512, 512, 512);
    k_attn<<<dim3(8192), dim3(256), 0, stream>>>(qf, kf, vf, attn16);
    gemm(attn16, ow16 + (size_t)l * 512 * 512, ob + l * 512, t0, 512, 512, 512, 512);
    k_add_ln<<<dim3(M_), dim3(256), 0, stream>>>(xf32, t0, g1 + l * 512, be1 + l * 512,
                                                 xf32, xf16);
    gemm(xf16, w116 + (size_t)l * 2048 * 512, b1 + l * 2048, zin, 2048, 2048, 512, 2048);
    k_gelu<<<dim3(65536), dim3(256), 0, stream>>>(zin, hm16, M_ * 2048);
    gemm(hm16, w216 + (size_t)l * 512 * 2048, b2 + l * 512, t0, 512, 512, 2048, 512);
    k_add_ln<<<dim3(M_), dim3(256), 0, stream>>>(xf32, t0, g2 + l * 512, be2 + l * 512,
                                                 xf32, xf16);
  }

  // ---- decoder + log_softmax (in place on d_out) ----
  gemm(xf16, dec16, decbp, out, NTOKP_, NTOK_, 512, NTOK_);
  k_logsoftmax<<<dim3(M_), dim3(256), 0, stream>>>(out);
}
